// ConvAttentionNetwork_33062658244811
// MI455X (gfx1250) — matrix operand order repaired and checked by reading, not compiled, not run on hardware
//
#include <hip/hip_runtime.h>
#include <stdint.h>

// ---------------- types for WMMA ----------------
typedef __attribute__((ext_vector_type(16))) __bf16 v16bf;
typedef __attribute__((ext_vector_type(8)))  __bf16 v8bf;
typedef __attribute__((ext_vector_type(8)))  float  v8f;

__device__ __forceinline__ v16bf frag16(const __bf16* p0, const __bf16* p1) {
    v8bf a = *(const v8bf*)p0;
    v8bf b = *(const v8bf*)p1;
    return __builtin_shufflevector(a, b, 0,1,2,3,4,5,6,7,8,9,10,11,12,13,14,15);
}

__device__ __forceinline__ v8f wmma_bf16(v16bf a, v16bf b, v8f c) {
    return __builtin_amdgcn_wmma_f32_16x16x32_bf16(false, a, false, b, (short)0, c, false, false);
}

// ---------------- problem constants ----------------
// V=30000 D=256 K1=64 K2=64 W1=18 W2=19 W3=2 LB=256 B=128 T=12
#define NV   30000
#define ND   256
#define NK1  64
#define NK2  64
#define NW1  18
#define NW2  19
#define NLB  256
#define NB   128
#define NT_  12
#define LPAD 292            // LB + W1+W2+W3-3
#define L1LEN 275           // 292-18+1
#define L2LEN 257           // 275-19+1
#define KC1  4608           // 256*18
#define KC2  1216           // 64*19

// ws offsets (bytes), each 256B aligned
#define OFF_EBF   0ull                         // 30000*256 bf16 = 15,360,000
#define OFF_EMB   15360000ull                  // 292*128*256 bf16 = 19,136,512
#define OFF_W1R   34496512ull                  // 64*4608 bf16 = 589,824
#define OFF_W2R   35086336ull                  // 64*1216 bf16 = 155,648
#define OFF_L1    35241984ull                  // 275*128*64 bf16 = 4,505,600
#define OFF_L2B   39747584ull                  // 128*257*64 f32 = 8,421,376
#define OFF_H     48168960ull                  // 11*128*64 f32 = 360,448
#define OFF_NHAT  48529408ull                  // 1408*256 bf16 = 720,896

// ---------------- prep kernels ----------------
__global__ void k_convE(const float* __restrict__ E, __bf16* __restrict__ Ebf) {
    size_t i = (size_t)blockIdx.x * blockDim.x + threadIdx.x;
    size_t n = (size_t)NV * ND;
    for (; i < n; i += (size_t)gridDim.x * blockDim.x)
        Ebf[i] = (__bf16)E[i];
}

// emb[l][b][d] = bf16(E[bodies[l][b]][d]),  l>=256 -> PAD_IDX=0
__global__ void k_gather_emb(const int* __restrict__ bodies, const float* __restrict__ E,
                             __bf16* __restrict__ emb) {
    int lb = blockIdx.x;                 // 0..292*128-1
    int l = lb / NB, b = lb % NB;
    int idx = (l < NLB) ? bodies[l * NB + b] : 0;
    float v = E[(size_t)idx * ND + threadIdx.x];
    emb[(size_t)lb * ND + threadIdx.x] = (__bf16)v;
}

// w1r[k1][w*256+d] = conv1_w[k1][d][w]
__global__ void k_prep_w1(const float* __restrict__ w1, __bf16* __restrict__ w1r) {
    int i = blockIdx.x * blockDim.x + threadIdx.x;
    if (i >= NK1 * KC1) return;
    int k1 = i / KC1, rem = i % KC1;
    int w = rem / ND, d = rem % ND;
    w1r[i] = (__bf16)w1[k1 * KC1 + d * NW1 + w];
}

// w2r[k2][w*64+c] = conv2_w[k2][c][w]
__global__ void k_prep_w2(const float* __restrict__ w2, __bf16* __restrict__ w2r) {
    int i = blockIdx.x * blockDim.x + threadIdx.x;
    if (i >= NK2 * KC2) return;
    int k2 = i / KC2, rem = i % KC2;
    int w = rem / NK2, c = rem % NK2;
    w2r[i] = (__bf16)w2[k2 * KC2 + c * NW2 + w];
}

// ---------------- conv1 as WMMA implicit GEMM ----------------
// M=64 (chan), K=4608 (w*256+d), N=128*275 (n=b*275+l). out: L1[l][b][chan] bf16, PReLU.
__global__ void k_conv1(const __bf16* __restrict__ emb, const __bf16* __restrict__ w1r,
                        const float* __restrict__ b1, const float* __restrict__ pa,
                        __bf16* __restrict__ L1) {
    const int wave = (blockIdx.x * blockDim.x + threadIdx.x) >> 5;
    const int lane = threadIdx.x & 31;
    const int NTIL = (NB * L1LEN) / 16;          // 2200
    const int mt = wave / NTIL;                  // 0..3
    const int nt = wave % NTIL;
    const int half = lane >> 4, r = lane & 15;
    const int arow = mt * 16 + r;
    const int ncol = nt * 16 + r;
    const int b = ncol / L1LEN, l = ncol % L1LEN;
    const __bf16* ap = w1r + (size_t)arow * KC1;
    v8f c = {};
    for (int k = 0; k < KC1; k += 32) {
        int ka = k + half * 16;
        v16bf af = frag16(ap + ka, ap + ka + 8);
        int kb = k + half * 16;                  // 16 contiguous K, same w (256%16==0)
        int w = kb >> 8, d = kb & 255;
        const __bf16* bp = emb + (size_t)(l + w) * (NB * ND) + b * ND + d;
        v16bf bf = frag16(bp, bp + 8);
        c = wmma_bf16(af, bf, c);
    }
    float alpha = *pa;
    int mbase = mt * 16 + 8 * half;
    v8bf ov;
#pragma unroll
    for (int i = 0; i < 8; i++) {
        float v = c[i] + b1[mbase + i];
        v = v >= 0.f ? v : alpha * v;
        ov[i] = (__bf16)v;
    }
    *(v8bf*)(L1 + (size_t)l * (NB * NK1) + b * NK1 + mbase) = ov;
}

// ---------------- conv2 as WMMA implicit GEMM ----------------
// M=64, K=1216 (w*64+c), N=128*257 (n=b*257+l). out: L2b[b][l][chan] f32 (+bias).
__global__ void k_conv2(const __bf16* __restrict__ L1, const __bf16* __restrict__ w2r,
                        const float* __restrict__ b2, float* __restrict__ L2b) {
    const int wave = (blockIdx.x * blockDim.x + threadIdx.x) >> 5;
    const int lane = threadIdx.x & 31;
    const int NTIL = (NB * L2LEN) / 16;          // 2056
    const int mt = wave / NTIL;
    const int nt = wave % NTIL;
    const int half = lane >> 4, r = lane & 15;
    const int arow = mt * 16 + r;
    const int ncol = nt * 16 + r;
    const int b = ncol / L2LEN, l = ncol % L2LEN;
    const __bf16* ap = w2r + (size_t)arow * KC2;
    v8f c = {};
    for (int k = 0; k < KC2; k += 32) {
        int ka = k + half * 16;
        v16bf af = frag16(ap + ka, ap + ka + 8);
        int kb = k + half * 16;                  // same w across 16 (64%16==0)
        int w = kb >> 6, cc = kb & 63;
        const __bf16* bp = L1 + (size_t)(l + w) * (NB * NK1) + b * NK1 + cc;
        v16bf bf = frag16(bp, bp + 8);
        c = wmma_bf16(af, bf, c);
    }
    int mbase = mt * 16 + 8 * half;
    float* dst = L2b + (size_t)b * (L2LEN * NK2) + l * NK2 + mbase;
#pragma unroll
    for (int i = 0; i < 8; i++)
        dst[i] = c[i] + b2[mbase + i];
}

// ---------------- GRU gates (h_prev = 0, so no recurrence) ----------------
__global__ void k_gru(const int* __restrict__ names, const float* __restrict__ E,
                      const float* __restrict__ wih, const float* __restrict__ bih,
                      const float* __restrict__ bhh, float* __restrict__ h) {
    __shared__ float xs[ND];
    int tb = blockIdx.x;                 // 0..1407
    int t = tb / NB, b = tb % NB;
    int name = names[t * NB + b];
    for (int d = threadIdx.x; d < ND; d += blockDim.x)
        xs[d] = E[(size_t)name * ND + d];
    __syncthreads();
    int k = threadIdx.x;
    float ar = 0.f, az = 0.f, an = 0.f;
    const float* wr = wih + (size_t)k * ND;
    const float* wz = wih + (size_t)(k + NK2) * ND;
    const float* wn = wih + (size_t)(k + 2 * NK2) * ND;
    for (int d = 0; d < ND; d++) {
        float x = xs[d];
        ar += x * wr[d]; az += x * wz[d]; an += x * wn[d];
    }
    ar += bih[k] + bhh[k];
    az += bih[k + NK2] + bhh[k + NK2];
    float rr = 1.f / (1.f + __expf(-ar));
    float zz = 1.f / (1.f + __expf(-az));
    float nn = tanhf(an + bih[k + 2 * NK2] + rr * bhh[k + 2 * NK2]);
    h[(size_t)tb * NK2 + k] = (1.f - zz) * nn;
}

// ---------------- per-(t,b): norm + conv3 + softmax + attention matvec ----------------
__global__ void k_attn(const int* __restrict__ bodies, const float* __restrict__ E,
                       const float* __restrict__ L2b, const float* __restrict__ hbuf,
                       const float* __restrict__ w3, const float* __restrict__ b3,
                       __bf16* __restrict__ nhat) {
    __shared__ float hs[NK2], w30[NK2], w31[NK2];
    __shared__ float q0[L2LEN], q1[L2LEN];
    __shared__ float red[256];
    __shared__ float alpha[NLB];
    __shared__ int   idx[NLB];
    int tb = blockIdx.x;                 // 0..1407
    int b = tb % NB;
    int tid = threadIdx.x;
    if (tid < NK2) {
        hs[tid]  = hbuf[(size_t)tb * NK2 + tid];
        w30[tid] = w3[tid * 2 + 0];
        w31[tid] = w3[tid * 2 + 1];
    }
    idx[tid] = bodies[tid * NB + b];
    __syncthreads();
    for (int l = tid; l < L2LEN; l += 256) {
        const float* p = L2b + (size_t)b * (L2LEN * NK2) + l * NK2;
        float s = 0.f, a0 = 0.f, a1 = 0.f;
        for (int k = 0; k < NK2; k++) {
            float v = p[k] * hs[k];
            s += v * v; a0 += v * w30[k]; a1 += v * w31[k];
        }
        float inv = 1.f / fmaxf(sqrtf(s), 1e-12f);
        q0[l] = a0 * inv;
        q1[l] = a1 * inv;
    }
    __syncthreads();
    float score = q0[tid] + q1[tid + 1] + b3[0];
    red[tid] = score;
    __syncthreads();
    for (int s = 128; s > 0; s >>= 1) {
        if (tid < s) red[tid] = fmaxf(red[tid], red[tid + s]);
        __syncthreads();
    }
    float mx = red[0];
    __syncthreads();
    float e = __expf(score - mx);
    red[tid] = e;
    __syncthreads();
    for (int s = 128; s > 0; s >>= 1) {
        if (tid < s) red[tid] += red[tid + s];
        __syncthreads();
    }
    alpha[tid] = e / red[0];
    __syncthreads();
    float acc = 0.f;                     // n_hat over d = tid
    for (int l = 0; l < NLB; l++)
        acc += alpha[l] * E[(size_t)idx[l] * ND + tid];
    nhat[(size_t)tb * ND + tid] = (__bf16)acc;
}

// ---------------- zero output slice t=0 ----------------
__global__ void k_zero(float* __restrict__ out) {
    size_t i = (size_t)blockIdx.x * blockDim.x + threadIdx.x;
    size_t n = (size_t)NB * NV;
    for (; i < n; i += (size_t)gridDim.x * blockDim.x) out[i] = 0.f;
}

// ---------------- logits GEMM: (1408x256) x (256x30000)^T via WMMA ----------------
// each wave: 1 M-tile x 3 N-tiles (1875 = 3*625), K loop 256/32 = 8.
__global__ void k_logits(const __bf16* __restrict__ Abf, const __bf16* __restrict__ Ebf,
                         const float* __restrict__ obias, float* __restrict__ out) {
    const int wave = (blockIdx.x * blockDim.x + threadIdx.x) >> 5;
    const int lane = threadIdx.x & 31;
    const int NG = 625;
    const int mt = wave / NG;
    const int ng = wave % NG;
    const int half = lane >> 4, r = lane & 15;
    const int arow = mt * 16 + r;
    const int n0 = (ng * 3 + 0) * 16 + r;
    const int n1 = (ng * 3 + 1) * 16 + r;
    const int n2 = (ng * 3 + 2) * 16 + r;
    const __bf16* ap  = Abf + (size_t)arow * ND;
    const __bf16* bp0 = Ebf + (size_t)n0 * ND;
    const __bf16* bp1 = Ebf + (size_t)n1 * ND;
    const __bf16* bp2 = Ebf + (size_t)n2 * ND;
    v8f c0 = {}, c1 = {}, c2 = {};
#pragma unroll
    for (int k = 0; k < ND; k += 32) {
        int ka = k + half * 16;
        int kb = k + half * 16;
        v16bf af = frag16(ap + ka, ap + ka + 8);
        v16bf b0 = frag16(bp0 + kb, bp0 + kb + 8);
        v16bf b1 = frag16(bp1 + kb, bp1 + kb + 8);
        v16bf b2 = frag16(bp2 + kb, bp2 + kb + 8);
        c0 = wmma_bf16(af, b0, c0);
        c1 = wmma_bf16(af, b1, c1);
        c2 = wmma_bf16(af, b2, c2);
    }
    int mbase = mt * 16 + 8 * half;
    float bb0 = obias[n0], bb1 = obias[n1], bb2 = obias[n2];
#pragma unroll
    for (int i = 0; i < 8; i++) {
        float* row = out + (size_t)(mbase + i + NB) * NV;   // +NB: slice t=0 is zeros
        row[n0] = c0[i] + bb0;
        row[n1] = c1[i] + bb1;
        row[n2] = c2[i] + bb2;
    }
}

// ---------------- launcher ----------------
extern "C" void kernel_launch(void* const* d_in, const int* in_sizes, int n_in,
                              void* d_out, int out_size, void* d_ws, size_t ws_size,
                              hipStream_t stream) {
    const int*   bodies = (const int*)  d_in[0];
    const int*   names  = (const int*)  d_in[1];
    const float* E      = (const float*)d_in[2];
    const float* c1w    = (const float*)d_in[3];
    const float* c1b    = (const float*)d_in[4];
    const float* c2w    = (const float*)d_in[5];
    const float* c2b    = (const float*)d_in[6];
    const float* c3w    = (const float*)d_in[7];
    const float* c3b    = (const float*)d_in[8];
    const float* pa     = (const float*)d_in[9];
    const float* wih    = (const float*)d_in[10];
    // d_in[11] = gru_w_hh unused (h_prev == 0)
    const float* bih    = (const float*)d_in[12];
    const float* bhh    = (const float*)d_in[13];
    const float* obias  = (const float*)d_in[14];
    float* out = (float*)d_out;

    char* ws = (char*)d_ws;
    __bf16* Ebf  = (__bf16*)(ws + OFF_EBF);
    __bf16* emb  = (__bf16*)(ws + OFF_EMB);
    __bf16* w1r  = (__bf16*)(ws + OFF_W1R);
    __bf16* w2r  = (__bf16*)(ws + OFF_W2R);
    __bf16* L1   = (__bf16*)(ws + OFF_L1);
    float*  L2b  = (float*) (ws + OFF_L2B);
    float*  h    = (float*) (ws + OFF_H);
    __bf16* nhat = (__bf16*)(ws + OFF_NHAT);

    // prep
    k_convE     <<<4096, 256, 0, stream>>>(E, Ebf);
    k_gather_emb<<<LPAD * NB, ND, 0, stream>>>(bodies, E, emb);
    k_prep_w1   <<<(NK1 * KC1 + 255) / 256, 256, 0, stream>>>(c1w, w1r);
    k_prep_w2   <<<(NK2 * KC2 + 255) / 256, 256, 0, stream>>>(c2w, w2r);

    // conv stack (WMMA)
    k_conv1<<<(4 * 2200) / 8, 256, 0, stream>>>(emb, w1r, c1b, pa, L1);
    k_conv2<<<(4 * 2056) / 8, 256, 0, stream>>>(L1, w2r, c2b, L2b);

    // gru gates + attention
    k_gru <<<(NT_ - 1) * NB, NK2, 0, stream>>>(names, E, wih, bih, bhh, h);
    k_attn<<<(NT_ - 1) * NB, 256, 0, stream>>>(bodies, E, L2b, h, c3w, c3b, nhat);

    // output
    k_zero  <<<4096, 256, 0, stream>>>(out);
    k_logits<<<(88 * 625) / 8, 256, 0, stream>>>(nhat, Ebf, obias, out);
}
